// _MACBlock_7473243095582
// MI455X (gfx1250) — compile-verified, matrix operand order checked by reading, not run on hardware
//
#include <hip/hip_runtime.h>
#include <math.h>

// ---------------- problem constants ----------------
#define Bc 2
#define Tc 4096
#define Dc 1024
#define Cc 512
#define Pc 16
#define Hc 2048
#define Fc 4096
#define Sc (Pc + 2 * Cc)   // 1040
#define NCH (Tc / Cc)      // 8

#define LRc    0.01f
#define BETAc  0.9f
#define ALPHAc 0.01f
#define EPSLN  1e-5f

typedef __attribute__((ext_vector_type(16))) __bf16 bf16x16;
typedef __attribute__((ext_vector_type(8)))  float  f32x8;

// ---------------- WMMA GEMM ----------------
// C[M,N] = A[M,K] @ B[K,N] (+bias[N]), A/B bf16, C f32.
// transA: A stored physically (K,M) row-major (lda = phys row stride)
// transB: B stored physically (N,K) row-major (ldb = phys row stride)
// Batched via blockIdx.z with element strides sA/sB/sC.
#define BM 128
#define BN 64
#define BK 32

union FragAB { bf16x16 v; uint4 q[2]; };
union Pack16 { __bf16 e[16]; uint4 q[2]; };
union Pack8  { __bf16 e[8];  uint4 q;    };

// gfx1250 async copy: LDS[lds_off + IOFF] = MEM[gaddr + IOFF], 16B per lane,
// tracked with ASYNCcnt (ISA 15.18.3 op 98; same IOFFSET applies to both sides).
__device__ __forceinline__ void async_g2l_b128(unsigned lds_off, unsigned long long gaddr) {
  asm volatile("global_load_async_to_lds_b128 %0, %1, off" :: "v"(lds_off), "v"(gaddr) : "memory");
}
__device__ __forceinline__ void async_g2l_b128_off16(unsigned lds_off, unsigned long long gaddr) {
  asm volatile("global_load_async_to_lds_b128 %0, %1, off offset:16" :: "v"(lds_off), "v"(gaddr) : "memory");
}
__device__ __forceinline__ void wait_async0() {
  asm volatile("s_wait_asynccnt 0x0" ::: "memory");
}
__device__ __forceinline__ unsigned lds_offset_of(const void* p) {
  // generic pointer to LDS: low 32 bits are the wave-relative LDS byte offset
  return (unsigned)(unsigned long long)p;
}

__global__ __launch_bounds__(256)
void wmma_gemm_kernel(const __bf16* __restrict__ A, const __bf16* __restrict__ B,
                      const float* __restrict__ bias, float* __restrict__ Cmat,
                      int M, int N, int K, int lda, int ldb, int ldc,
                      int transA, int transB,
                      long long sA, long long sB, long long sC)
{
  __shared__ __align__(16) __bf16 As[BM * BK];
  __shared__ __align__(16) __bf16 Bs[BN * BK];   // stored Bs[n][k]

  const int tid  = threadIdx.x;
  const int wave = tid >> 5, lane = tid & 31;
  const int wm = wave & 3, wn = wave >> 2;       // 4x2 wave grid -> 128x64 tile
  const int lmod = lane & 15, lhalf = lane >> 4;
  const long long zb = blockIdx.z;
  A += sA * zb; B += sB * zb; Cmat += sC * zb;
  const int m0 = blockIdx.y * BM, n0 = blockIdx.x * BN;

  f32x8 acc[2][2];
  for (int a = 0; a < 2; ++a)
    for (int b2 = 0; b2 < 2; ++b2)
      for (int i = 0; i < 8; ++i) acc[a][b2][i] = 0.0f;

  for (int k0 = 0; k0 < K; k0 += BK) {
    // ---- stage A tile As[r][c] = A[m0+r, k0+c] ----
    if (!transA) {
      int r = tid >> 1, c0 = (tid & 1) * 16;
      int gr = m0 + r;
      unsigned lds = lds_offset_of(&As[r * BK + c0]);
      if (gr < M && (k0 + c0 + 16) <= K) {
        unsigned long long ga = (unsigned long long)(A + (long long)gr * lda + k0 + c0);
        async_g2l_b128(lds, ga);           // 16B -> LDS, no VGPR round-trip
        async_g2l_b128_off16(lds, ga);     // next 16B (offset applies to both sides)
      } else {
        uint4* q = reinterpret_cast<uint4*>(&As[r * BK + c0]);
        q[0] = make_uint4(0, 0, 0, 0); q[1] = make_uint4(0, 0, 0, 0);
      }
    } else {
      int c = tid >> 3, r0 = (tid & 7) * 16;
      int gk = k0 + c;
      Pack16 t; t.q[0] = make_uint4(0, 0, 0, 0); t.q[1] = make_uint4(0, 0, 0, 0);
      if (gk < K && (m0 + r0 + 16) <= M) {
        const uint4* p = reinterpret_cast<const uint4*>(A + (long long)gk * lda + m0 + r0);
        t.q[0] = p[0]; t.q[1] = p[1];
        if (k0 + BK < K) __builtin_prefetch(A + (long long)(gk + BK) * lda + m0 + r0, 0, 0);
      }
      #pragma unroll
      for (int i = 0; i < 16; ++i) As[(r0 + i) * BK + c] = t.e[i];
    }
    // ---- stage B tile Bs[n][c] = B[k0+c, n0+n] ----
    if (!transB) {
      int c = tid >> 3, nn0 = (tid & 7) * 8;
      int gk = k0 + c;
      Pack8 t; t.q = make_uint4(0, 0, 0, 0);
      if (gk < K && (n0 + nn0 + 8) <= N) {
        t.q = *reinterpret_cast<const uint4*>(B + (long long)gk * ldb + n0 + nn0);
        if (k0 + BK < K) __builtin_prefetch(B + (long long)(gk + BK) * ldb + n0 + nn0, 0, 0);
      }
      #pragma unroll
      for (int i = 0; i < 8; ++i) Bs[(nn0 + i) * BK + c] = t.e[i];
    } else {
      int n = tid >> 2, c0 = (tid & 3) * 8;
      int gn = n0 + n;
      unsigned lds = lds_offset_of(&Bs[n * BK + c0]);
      if (gn < N && (k0 + c0 + 8) <= K) {
        unsigned long long ga = (unsigned long long)(B + (long long)gn * ldb + k0 + c0);
        async_g2l_b128(lds, ga);
      } else {
        *reinterpret_cast<uint4*>(&Bs[n * BK + c0]) = make_uint4(0, 0, 0, 0);
      }
    }
    wait_async0();          // ASYNCcnt: our async transfers landed in LDS
    __syncthreads();        // DScnt + barrier: scatter stores from all waves visible

    // ---- fragments (ISA 16-bit A 16x32 layout: lanes 0-15 K=0..7/16..23, 16-31 K=8..15/24..31)
    FragAB fa[2], fb[2];
    #pragma unroll
    for (int tr = 0; tr < 2; ++tr) {
      int am = wm * 32 + tr * 16 + lmod;
      fa[tr].q[0] = *reinterpret_cast<const uint4*>(&As[am * BK + lhalf * 8]);
      fa[tr].q[1] = *reinterpret_cast<const uint4*>(&As[am * BK + 16 + lhalf * 8]);
    }
    #pragma unroll
    for (int tn = 0; tn < 2; ++tn) {
      int bn = wn * 32 + tn * 16 + lmod;
      fb[tn].q[0] = *reinterpret_cast<const uint4*>(&Bs[bn * BK + lhalf * 8]);
      fb[tn].q[1] = *reinterpret_cast<const uint4*>(&Bs[bn * BK + 16 + lhalf * 8]);
    }
    #pragma unroll
    for (int tr = 0; tr < 2; ++tr)
      #pragma unroll
      for (int tn = 0; tn < 2; ++tn)
        acc[tr][tn] = __builtin_amdgcn_wmma_f32_16x16x32_bf16(
            false, fa[tr].v, false, fb[tn].v, (short)0, acc[tr][tn], false, false);
    __syncthreads();
  }

  // ---- store (C layout: lane n=lmod, VGPR i -> row i + 8*lhalf) ----
  #pragma unroll
  for (int tr = 0; tr < 2; ++tr) {
    #pragma unroll
    for (int tn = 0; tn < 2; ++tn) {
      int col = n0 + wn * 32 + tn * 16 + lmod;
      if (col >= N) continue;
      float bv = bias ? bias[col] : 0.0f;
      #pragma unroll
      for (int i = 0; i < 8; ++i) {
        int row = m0 + wm * 32 + tr * 16 + lhalf * 8 + i;
        if (row < M) Cmat[(long long)row * ldc + col] = acc[tr][tn][i] + bv;
      }
    }
  }
}

// ---------------- elementwise / reduction kernels ----------------
__global__ void cast_f32_bf16_k(const float* __restrict__ in, __bf16* __restrict__ out, long long n) {
  long long i = (long long)blockIdx.x * 256 + threadIdx.x;
  if (i < n) out[i] = (__bf16)in[i];
}

__global__ void silu_bf16_k(const float* __restrict__ z, __bf16* __restrict__ out, long long n) {
  long long i = (long long)blockIdx.x * 256 + threadIdx.x;
  if (i < n) { float v = z[i]; float s = 1.0f / (1.0f + expf(-v)); out[i] = (__bf16)(v * s); }
}

__global__ void dpred_k(const float* __restrict__ pred, const float* __restrict__ vv,
                        __bf16* __restrict__ out, long long n, float inv) {
  long long i = (long long)blockIdx.x * 256 + threadIdx.x;
  if (i < n) out[i] = (__bf16)((pred[i] - vv[i]) * inv);
}

__global__ void dsilu_k(const float* __restrict__ dh, const float* __restrict__ z,
                        __bf16* __restrict__ out, long long n) {
  long long i = (long long)blockIdx.x * 256 + threadIdx.x;
  if (i < n) {
    float zv = z[i]; float s = 1.0f / (1.0f + expf(-zv));
    out[i] = (__bf16)(dh[i] * s * (1.0f + zv * (1.0f - s)));
  }
}

__global__ void update_k(float* __restrict__ mem, float* __restrict__ mom,
                         const float* __restrict__ g, __bf16* __restrict__ bfout, long long n) {
  long long i = (long long)blockIdx.x * 256 + threadIdx.x;
  if (i < n) {
    float mo = BETAc * mom[i] + g[i]; mom[i] = mo;
    float mv = (1.0f - ALPHAc) * mem[i] - LRc * mo; mem[i] = mv;
    if (bfout) bfout[i] = (__bf16)mv;
  }
}

__global__ void pack_seg_k(const float* __restrict__ x, __bf16* __restrict__ out, int t0) {
  long long n = (long long)Bc * Cc * Dc;
  long long i = (long long)blockIdx.x * 256 + threadIdx.x;
  if (i < n) {
    long long b = i / ((long long)Cc * Dc), rem = i % ((long long)Cc * Dc);
    out[i] = (__bf16)x[b * Tc * Dc + (long long)t0 * Dc + rem];
  }
}

__global__ void build_aug_k(const float* __restrict__ x, const float* __restrict__ hm,
                            const float* __restrict__ pers, float* __restrict__ aug, int t0) {
  long long n = (long long)Bc * Sc * Dc;
  long long i = (long long)blockIdx.x * 256 + threadIdx.x;
  if (i < n) {
    int d = (int)(i % Dc); int s = (int)((i / Dc) % Sc); long long b = i / ((long long)Dc * Sc);
    float v;
    if (s < Pc)           v = pers[s * Dc + d];
    else if (s < Pc + Cc) v = hm[(b * Cc + (s - Pc)) * Dc + d];
    else                  v = x[(b * Tc + t0 + (s - Pc - Cc)) * (long long)Dc + d];
    aug[i] = v;
  }
}

__global__ void add_seg_k(const float* __restrict__ x, const float* __restrict__ attn,
                          float* __restrict__ y, int t0) {
  long long n = (long long)Bc * Cc * Dc;
  long long i = (long long)blockIdx.x * 256 + threadIdx.x;
  if (i < n) {
    int d = (int)(i % Dc); int c = (int)((i / Dc) % Cc); long long b = i / ((long long)Dc * Cc);
    y[i] = x[(b * Tc + t0 + c) * (long long)Dc + d] + attn[i];
  }
}

__global__ void gate_combine_k(const float* __restrict__ y, const float* __restrict__ m,
                               const float* __restrict__ gl, float* __restrict__ o, long long n) {
  long long i = (long long)blockIdx.x * 256 + threadIdx.x;
  if (i < n) { float s = 1.0f / (1.0f + expf(-gl[i])); o[i] = y[i] * s + m[i] * (1.0f - s); }
}

__global__ void final_out_k(const float* __restrict__ o, const float* __restrict__ f2,
                            float* __restrict__ out, int t0) {
  long long n = (long long)Bc * Cc * Dc;
  long long i = (long long)blockIdx.x * 256 + threadIdx.x;
  if (i < n) {
    int d = (int)(i % Dc); int c = (int)((i / Dc) % Cc); long long b = i / ((long long)Dc * Cc);
    out[(b * Tc + t0 + c) * (long long)Dc + d] = o[i] + f2[i];
  }
}

__global__ void l2norm_k(const float* __restrict__ in, __bf16* __restrict__ out, float scale) {
  const float* row = in + (long long)blockIdx.x * Dc;
  __bf16* orow = out + (long long)blockIdx.x * Dc;
  __shared__ float red[256];
  float s = 0.0f;
  for (int i = threadIdx.x; i < Dc; i += 256) { float v = row[i]; s += v * v; }
  red[threadIdx.x] = s; __syncthreads();
  for (int st = 128; st > 0; st >>= 1) { if (threadIdx.x < st) red[threadIdx.x] += red[threadIdx.x + st]; __syncthreads(); }
  float inv = scale / fmaxf(sqrtf(red[0]), 1e-12f);
  for (int i = threadIdx.x; i < Dc; i += 256) orow[i] = (__bf16)(row[i] * inv);
}

__global__ void layernorm_k(const float* __restrict__ in, const float* __restrict__ g,
                            const float* __restrict__ b, __bf16* __restrict__ out) {
  const float* row = in + (long long)blockIdx.x * Dc;
  __bf16* orow = out + (long long)blockIdx.x * Dc;
  __shared__ float r1[256]; __shared__ float r2[256];
  float s = 0.0f, s2 = 0.0f;
  for (int i = threadIdx.x; i < Dc; i += 256) { float v = row[i]; s += v; s2 += v * v; }
  r1[threadIdx.x] = s; r2[threadIdx.x] = s2; __syncthreads();
  for (int st = 128; st > 0; st >>= 1) {
    if (threadIdx.x < st) { r1[threadIdx.x] += r1[threadIdx.x + st]; r2[threadIdx.x] += r2[threadIdx.x + st]; }
    __syncthreads();
  }
  float m = r1[0] / Dc; float var = r2[0] / Dc - m * m;
  float inv = rsqrtf(var + EPSLN);
  for (int i = threadIdx.x; i < Dc; i += 256)
    orow[i] = (__bf16)((row[i] - m) * inv * g[i] + b[i]);
}

__global__ void softmax_mask_k(const float* __restrict__ scores, __bf16* __restrict__ probs) {
  int r = blockIdx.x;                 // 0 .. B*C-1 (query rows P+C..S-1)
  int i = r % Cc;
  int limit = Pc + Cc + i + 1;        // allowed cols: j <= P+C+i
  const float* srow = scores + (long long)r * Sc;
  __bf16* prow = probs + (long long)r * Sc;
  __shared__ float red[256];
  float mx = -1e30f;
  for (int j = threadIdx.x; j < limit; j += 256) mx = fmaxf(mx, srow[j]);
  red[threadIdx.x] = mx; __syncthreads();
  for (int st = 128; st > 0; st >>= 1) { if (threadIdx.x < st) red[threadIdx.x] = fmaxf(red[threadIdx.x], red[threadIdx.x + st]); __syncthreads(); }
  mx = red[0]; __syncthreads();
  float s = 0.0f;
  for (int j = threadIdx.x; j < limit; j += 256) s += expf(srow[j] - mx);
  red[threadIdx.x] = s; __syncthreads();
  for (int st = 128; st > 0; st >>= 1) { if (threadIdx.x < st) red[threadIdx.x] += red[threadIdx.x + st]; __syncthreads(); }
  float inv = 1.0f / red[0];
  for (int j = threadIdx.x; j < Sc; j += 256)
    prow[j] = (__bf16)((j < limit) ? expf(srow[j] - mx) * inv : 0.0f);
}

__global__ void colsum_k(const __bf16* __restrict__ A, float* __restrict__ out, int Mrows, int Ncols) {
  int col = blockIdx.x;
  __shared__ float red[256];
  float s = 0.0f;
  for (int r = threadIdx.x; r < Mrows; r += 256) s += (float)A[(long long)r * Ncols + col];
  red[threadIdx.x] = s; __syncthreads();
  for (int st = 128; st > 0; st >>= 1) { if (threadIdx.x < st) red[threadIdx.x] += red[threadIdx.x + st]; __syncthreads(); }
  if (threadIdx.x == 0) out[col] = red[0];
}

// ---------------- host orchestration ----------------
extern "C" void kernel_launch(void* const* d_in, const int* in_sizes, int n_in,
                              void* d_out, int out_size, void* d_ws, size_t ws_size,
                              hipStream_t stream) {
  (void)in_sizes; (void)n_in; (void)out_size; (void)ws_size;
  const float* x      = (const float*)d_in[0];
  const float* wq_ctx = (const float*)d_in[1];
  const float* wq     = (const float*)d_in[2];
  const float* wk     = (const float*)d_in[3];
  const float* wv     = (const float*)d_in[4];
  const float* w_attn = (const float*)d_in[5];
  const float* w_gate = (const float*)d_in[6];
  const float* gate_g = (const float*)d_in[7];
  const float* gate_b = (const float*)d_in[8];
  const float* n1_g   = (const float*)d_in[9];
  const float* n1_b   = (const float*)d_in[10];
  const float* n2_g   = (const float*)d_in[11];
  const float* n2_b   = (const float*)d_in[12];
  const float* ffn_w1 = (const float*)d_in[13];
  const float* ffn_b1 = (const float*)d_in[14];
  const float* ffn_w2 = (const float*)d_in[15];
  const float* ffn_b2 = (const float*)d_in[16];
  const float* pers   = (const float*)d_in[17];
  const float* mw0_in = (const float*)d_in[18];
  const float* mb0_in = (const float*)d_in[19];
  const float* mw1_in = (const float*)d_in[20];
  const float* mb1_in = (const float*)d_in[21];
  const float* mem_wk = (const float*)d_in[22];
  const float* mem_wv = (const float*)d_in[23];
  float* out = (float*)d_out;

  const long long DD  = (long long)Dc * Dc;
  const long long DH  = (long long)Dc * Hc;
  const long long DF  = (long long)Dc * Fc;
  const long long BSD = (long long)Bc * Sc * Dc;
  const long long BCD = (long long)Bc * Cc * Dc;
  const long long BCS = (long long)Bc * Cc * Sc;

  char* base = (char*)d_ws;
  size_t off = 0;
  auto alloc = [&](size_t bytes) -> void* {
    void* p = base + off; off = (off + bytes + 255) & ~(size_t)255; return p;
  };
  // bf16 weights
  __bf16* wb_qctx = (__bf16*)alloc(DD * 2);
  __bf16* wb_q    = (__bf16*)alloc(DD * 2);
  __bf16* wb_k    = (__bf16*)alloc(DD * 2);
  __bf16* wb_v    = (__bf16*)alloc(DD * 2);
  __bf16* wb_attn = (__bf16*)alloc(DD * 2);
  __bf16* wb_gate = (__bf16*)alloc(DD * 2);
  __bf16* wb_mk   = (__bf16*)alloc(DD * 2);
  __bf16* wb_mv   = (__bf16*)alloc(DD * 2);
  __bf16* wb_f1   = (__bf16*)alloc(DF * 2);
  __bf16* wb_f2   = (__bf16*)alloc(DF * 2);
  __bf16* mw0_bf  = (__bf16*)alloc(DH * 2);
  __bf16* mw1_bf  = (__bf16*)alloc(DH * 2);
  // f32 memory state + momentum
  float* mw0 = (float*)alloc(DH * 4);  float* mb0 = (float*)alloc(Hc * 4);
  float* mw1 = (float*)alloc(DH * 4);  float* mb1 = (float*)alloc(Dc * 4);
  float* ow0 = (float*)alloc(DH * 4);  float* ob0 = (float*)alloc(Hc * 4);
  float* ow1 = (float*)alloc(DH * 4);  float* ob1 = (float*)alloc(Dc * 4);
  // bf16 temporaries
  __bf16* seg_bf   = (__bf16*)alloc(BCD * 2);
  __bf16* segq_bf  = (__bf16*)alloc(BCD * 2);
  __bf16* lnaug_bf = (__bf16*)alloc(BSD * 2);
  __bf16* k_bf     = (__bf16*)alloc(BSD * 2);
  __bf16* v_bf     = (__bf16*)alloc(BSD * 2);
  __bf16* q_bf     = (__bf16*)alloc(BCD * 2);
  __bf16* probs_bf = (__bf16*)alloc(BCS * 2);
  __bf16* attn1_bf = (__bf16*)alloc(BCD * 2);
  __bf16* yt_bf    = (__bf16*)alloc(BCD * 2);
  __bf16* kk_bf    = (__bf16*)alloc(BCD * 2);
  __bf16* h_bf     = (__bf16*)alloc(DH * 2);
  __bf16* dpred_bf = (__bf16*)alloc(BCD * 2);
  __bf16* dz0_bf   = (__bf16*)alloc(DH * 2);
  __bf16* lnyt_bf  = (__bf16*)alloc(BCD * 2);
  __bf16* lno_bf   = (__bf16*)alloc(BCD * 2);
  __bf16* fh_bf    = (__bf16*)alloc(DF * 2);
  // f32 temporaries
  float* lin1    = (float*)alloc(BCD * 4);
  float* z0      = (float*)alloc(DH * 4);
  float* mlp_out = (float*)alloc(BCD * 4);   // hmem / pred / mem_out (disjoint lifetimes)
  float* aug     = (float*)alloc(BSD * 4);
  float* klin    = (float*)alloc(BSD * 4);
  float* vlin    = (float*)alloc(BSD * 4);
  float* qlin    = (float*)alloc(BCD * 4);
  float* scores  = (float*)alloc(BCS * 4);
  float* attn1   = (float*)alloc(BCD * 4);
  float* attnout = (float*)alloc(BCD * 4);
  float* y_t     = (float*)alloc(BCD * 4);
  float* kk_lin  = (float*)alloc(BCD * 4);
  float* vvb     = (float*)alloc(BCD * 4);
  float* dh      = (float*)alloc(DH * 4);
  float* g_w1    = (float*)alloc(DH * 4);
  float* g_w0    = (float*)alloc(DH * 4);
  float* g_b0    = (float*)alloc(Hc * 4);
  float* g_b1    = (float*)alloc(Dc * 4);
  float* gl      = (float*)alloc(BCD * 4);
  float* o_buf   = (float*)alloc(BCD * 4);
  float* fz      = (float*)alloc(DF * 4);
  float* f2      = (float*)alloc(BCD * 4);

  auto blk = [](long long n) { return dim3((unsigned)((n + 255) / 256)); };
  auto cast = [&](const float* s, __bf16* d, long long n) {
    cast_f32_bf16_k<<<blk(n), 256, 0, stream>>>(s, d, n);
  };
  auto gemm = [&](const __bf16* Ap, const __bf16* Bp, const float* bias, float* Cp,
                  int M, int N, int K, int lda, int ldb, int ldc, int tA, int tB,
                  long long sA, long long sB, long long sC, int batch) {
    dim3 g((N + BN - 1) / BN, (M + BM - 1) / BM, batch);
    wmma_gemm_kernel<<<g, 256, 0, stream>>>(Ap, Bp, bias, Cp, M, N, K, lda, ldb, ldc, tA, tB, sA, sB, sC);
  };

  // ---- once-per-call setup (deterministic, capture-safe) ----
  cast(wq_ctx, wb_qctx, DD); cast(wq, wb_q, DD); cast(wk, wb_k, DD); cast(wv, wb_v, DD);
  cast(w_attn, wb_attn, DD); cast(w_gate, wb_gate, DD); cast(mem_wk, wb_mk, DD); cast(mem_wv, wb_mv, DD);
  cast(ffn_w1, wb_f1, DF); cast(ffn_w2, wb_f2, DF);
  hipMemcpyAsync(mw0, mw0_in, DH * 4, hipMemcpyDeviceToDevice, stream);
  hipMemcpyAsync(mb0, mb0_in, Hc * 4, hipMemcpyDeviceToDevice, stream);
  hipMemcpyAsync(mw1, mw1_in, DH * 4, hipMemcpyDeviceToDevice, stream);
  hipMemcpyAsync(mb1, mb1_in, Dc * 4, hipMemcpyDeviceToDevice, stream);
  hipMemsetAsync(ow0, 0, DH * 4, stream); hipMemsetAsync(ob0, 0, Hc * 4, stream);
  hipMemsetAsync(ow1, 0, DH * 4, stream); hipMemsetAsync(ob1, 0, Dc * 4, stream);
  cast(mw0, mw0_bf, DH); cast(mw1, mw1_bf, DH);

  const int BC = Bc * Cc;          // 1024 rows
  const int BS = Bc * Sc;          // 2080 rows
  const float scaleq = 32.0f;      // sqrt(D)
  const float invBC = 1.0f / (float)BC;

  for (int ch = 0; ch < NCH; ++ch) {
    const int t0 = ch * Cc;
    // -- memory read path: h = mem_mlp(l2norm(seg @ wq_ctx))
    pack_seg_k<<<blk(BCD), 256, 0, stream>>>(x, seg_bf, t0);
    gemm(seg_bf, wb_qctx, nullptr, lin1, BC, Dc, Dc, Dc, Dc, Dc, 0, 0, 0, 0, 0, 1);
    l2norm_k<<<BC, 256, 0, stream>>>(lin1, segq_bf, 1.0f);
    gemm(segq_bf, mw0_bf, mb0, z0, BC, Hc, Dc, Dc, Hc, Hc, 0, 0, 0, 0, 0, 1);
    silu_bf16_k<<<blk(DH), 256, 0, stream>>>(z0, h_bf, DH);
    gemm(h_bf, mw1_bf, mb1, mlp_out, BC, Dc, Hc, Hc, Dc, Dc, 0, 0, 0, 0, 0, 1);
    // -- aug = [pers | h | seg]; ln; k,v over all S rows; q over last C rows
    build_aug_k<<<blk(BSD), 256, 0, stream>>>(x, mlp_out, pers, aug, t0);
    layernorm_k<<<BS, 256, 0, stream>>>(aug, n1_g, n1_b, lnaug_bf);
    gemm(lnaug_bf, wb_k, nullptr, klin, BS, Dc, Dc, Dc, Dc, Dc, 0, 0, 0, 0, 0, 1);
    l2norm_k<<<BS, 256, 0, stream>>>(klin, k_bf, 1.0f);
    gemm(lnaug_bf, wb_v, nullptr, vlin, BS, Dc, Dc, Dc, Dc, Dc, 0, 0, 0, 0, 0, 1);
    cast(vlin, v_bf, BSD);
    gemm(lnaug_bf + (long long)(Pc + Cc) * Dc, wb_q, nullptr, qlin,
         Cc, Dc, Dc, Dc, Dc, Dc, 0, 0, (long long)Sc * Dc, 0, (long long)Cc * Dc, Bc);
    l2norm_k<<<BC, 256, 0, stream>>>(qlin, q_bf, scaleq);
    // -- scores = q @ k^T (batched, NT), masked softmax, attn = probs @ v, out proj
    gemm(q_bf, k_bf, nullptr, scores, Cc, Sc, Dc, Dc, Dc, Sc, 0, 1,
         (long long)Cc * Dc, (long long)Sc * Dc, (long long)Cc * Sc, Bc);
    softmax_mask_k<<<BC, 256, 0, stream>>>(scores, probs_bf);
    gemm(probs_bf, v_bf, nullptr, attn1, Cc, Dc, Sc, Sc, Dc, Dc, 0, 0,
         (long long)Cc * Sc, (long long)Sc * Dc, (long long)Cc * Dc, Bc);
    cast(attn1, attn1_bf, BCD);
    gemm(attn1_bf, wb_attn, nullptr, attnout, BC, Dc, Dc, Dc, Dc, Dc, 0, 0, 0, 0, 0, 1);
    add_seg_k<<<blk(BCD), 256, 0, stream>>>(x, attnout, y_t, t0);
    cast(y_t, yt_bf, BCD);
    // -- memory write path: kk, vv; MLP fwd+bwd; momentum update
    gemm(yt_bf, wb_mk, nullptr, kk_lin, BC, Dc, Dc, Dc, Dc, Dc, 0, 0, 0, 0, 0, 1);
    cast(kk_lin, kk_bf, BCD);
    gemm(yt_bf, wb_mv, nullptr, vvb, BC, Dc, Dc, Dc, Dc, Dc, 0, 0, 0, 0, 0, 1);
    gemm(kk_bf, mw0_bf, mb0, z0, BC, Hc, Dc, Dc, Hc, Hc, 0, 0, 0, 0, 0, 1);
    silu_bf16_k<<<blk(DH), 256, 0, stream>>>(z0, h_bf, DH);
    gemm(h_bf, mw1_bf, mb1, mlp_out, BC, Dc, Hc, Hc, Dc, Dc, 0, 0, 0, 0, 0, 1);
    dpred_k<<<blk(BCD), 256, 0, stream>>>(mlp_out, vvb, dpred_bf, BCD, invBC);
    colsum_k<<<Dc, 256, 0, stream>>>(dpred_bf, g_b1, BC, Dc);
    gemm(h_bf, dpred_bf, nullptr, g_w1, Hc, Dc, BC, Hc, Dc, Dc, 1, 0, 0, 0, 0, 1);   // h^T @ dpred
    gemm(dpred_bf, mw1_bf, nullptr, dh, BC, Hc, Dc, Dc, Dc, Hc, 0, 1, 0, 0, 0, 1);   // dpred @ w1^T
    dsilu_k<<<blk(DH), 256, 0, stream>>>(dh, z0, dz0_bf, DH);
    colsum_k<<<Hc, 256, 0, stream>>>(dz0_bf, g_b0, BC, Hc);
    gemm(kk_bf, dz0_bf, nullptr, g_w0, Dc, Hc, BC, Dc, Hc, Hc, 1, 0, 0, 0, 0, 1);    // kk^T @ dz0
    update_k<<<blk(DH), 256, 0, stream>>>(mw0, ow0, g_w0, mw0_bf, DH);
    update_k<<<blk(Hc), 256, 0, stream>>>(mb0, ob0, g_b0, nullptr, Hc);
    update_k<<<blk(DH), 256, 0, stream>>>(mw1, ow1, g_w1, mw1_bf, DH);
    update_k<<<blk(Dc), 256, 0, stream>>>(mb1, ob1, g_b1, nullptr, Dc);
    // -- mem_out = mem_mlp(mem_new, l2norm(y_t @ wq_ctx))
    gemm(yt_bf, wb_qctx, nullptr, lin1, BC, Dc, Dc, Dc, Dc, Dc, 0, 0, 0, 0, 0, 1);
    l2norm_k<<<BC, 256, 0, stream>>>(lin1, segq_bf, 1.0f);
    gemm(segq_bf, mw0_bf, mb0, z0, BC, Hc, Dc, Dc, Hc, Hc, 0, 0, 0, 0, 0, 1);
    silu_bf16_k<<<blk(DH), 256, 0, stream>>>(z0, h_bf, DH);
    gemm(h_bf, mw1_bf, mb1, mlp_out, BC, Dc, Hc, Hc, Dc, Dc, 0, 0, 0, 0, 0, 1);
    // -- gate + FFN + residual out
    layernorm_k<<<BC, 256, 0, stream>>>(y_t, gate_g, gate_b, lnyt_bf);
    gemm(lnyt_bf, wb_gate, nullptr, gl, BC, Dc, Dc, Dc, Dc, Dc, 0, 0, 0, 0, 0, 1);
    gate_combine_k<<<blk(BCD), 256, 0, stream>>>(y_t, mlp_out, gl, o_buf, BCD);
    layernorm_k<<<BC, 256, 0, stream>>>(o_buf, n2_g, n2_b, lno_bf);
    gemm(lno_bf, wb_f1, ffn_b1, fz, BC, Fc, Dc, Dc, Fc, Fc, 0, 0, 0, 0, 0, 1);
    silu_bf16_k<<<blk(DF), 256, 0, stream>>>(fz, fh_bf, DF);
    gemm(fh_bf, wb_f2, ffn_b2, f2, BC, Dc, Fc, Fc, Dc, Dc, 0, 0, 0, 0, 0, 1);
    final_out_k<<<blk(BCD), 256, 0, stream>>>(o_buf, f2, out, t0);
  }
}
